// FusionGCN_36567351558152
// MI455X (gfx1250) — matrix operand order repaired and checked by reading, not compiled, not run on hardware
//
#include <hip/hip_runtime.h>
#include <stdint.h>

#define BSZ   16
#define NPIX  1024
#define CS    256
#define CXC   64
#define CIN   320
#define PHIH  64
#define GHID  128
#define BN_EPS 1e-5f

typedef __bf16 bf16;
typedef __attribute__((ext_vector_type(16))) __bf16 v16bf;
typedef __attribute__((ext_vector_type(8)))  __bf16 v8bf;
typedef __attribute__((ext_vector_type(8)))  float  v8f;

static __device__ __forceinline__ v16bf cat8(v8bf lo, v8bf hi) {
  v16bf r;
#pragma unroll
  for (int i = 0; i < 8; ++i) { r[i] = lo[i]; r[i + 8] = hi[i]; }
  return r;
}

// ---------------------------------------------------------------- pack feat
// nf[b][n][c] (bf16) = concat(search, xcorr)[b][c][n]
__global__ __launch_bounds__(256) void k_pack(const float* __restrict__ sf,
                                              const float* __restrict__ xc,
                                              bf16* __restrict__ nf) {
  int idx = blockIdx.x * 256 + threadIdx.x;
  const int total = BSZ * NPIX * CIN;
  if (idx >= total) return;
  int c = idx % CIN;
  int n = (idx / CIN) % NPIX;
  int b = idx / (CIN * NPIX);
  float v = (c < CS) ? sf[((size_t)b * CS + c) * NPIX + n]
                     : xc[((size_t)b * CXC + (c - CS)) * NPIX + n];
  nf[idx] = (bf16)v;
}

// ------------------------------------------------------- weight transposes
// w1t[n][k] = phi_w1[k][n] (64x320);  g1t[n][k] = gc1_w[k][n] (128x320)
__global__ __launch_bounds__(256) void k_weights(const float* __restrict__ w1,
                                                 const float* __restrict__ g1,
                                                 bf16* __restrict__ w1t,
                                                 bf16* __restrict__ g1t) {
  int idx = blockIdx.x * 256 + threadIdx.x;
  if (idx < PHIH * CIN) {
    int n = idx / CIN, k = idx % CIN;
    w1t[idx] = (bf16)w1[k * PHIH + n];
  } else if (idx < PHIH * CIN + GHID * CIN) {
    int j = idx - PHIH * CIN;
    int n = j / CIN, k = j % CIN;
    g1t[j] = (bf16)g1[k * GHID + n];
  }
}

// ------------------------------------------- edge MLP GEMM: diff @ W1 + b1
// block: 256 thr (8 waves); 256 edges x 64 outputs; each wave owns 32 edges
// (two 16x32 A tiles sharing each B fragment). W1^T cached in LDS.
__global__ __launch_bounds__(256) void k_phi(const bf16* __restrict__ nf,
                                             const bf16* __restrict__ w1t,
                                             const int* __restrict__ pairs,
                                             const float* __restrict__ b1,
                                             float* __restrict__ hpre, int E) {
  __shared__ bf16 sW[PHIH][CIN];   // 40 KB  (sW[n][k], contiguous along K)
  __shared__ bf16 sD[256][32];     // 16 KB  current K-chunk of |f0-f1|
  __shared__ int  sP[256][2];      //  2 KB
  const int tid = threadIdx.x;
  const int b   = blockIdx.y;
  const int e0  = blockIdx.x * 256;

  __builtin_prefetch(w1t, 0, 1);   // global_prefetch_b8 (weight panel)
  {
    const uint32_t* src = (const uint32_t*)w1t;
    uint32_t* dst = (uint32_t*)&sW[0][0];
    for (int i = tid; i < PHIH * CIN / 2; i += 256) dst[i] = src[i];
  }
  {
    int e = e0 + tid; if (e >= E) e = E - 1;
    sP[tid][0] = pairs[((size_t)b * E + e) * 2 + 0];
    sP[tid][1] = pairs[((size_t)b * E + e) * 2 + 1];
  }

  const int lane = tid & 31, wv = tid >> 5;
  const int mrow = lane & 15, kh = lane >> 4;
  v8f acc[2][4];
#pragma unroll
  for (int t = 0; t < 2; ++t)
#pragma unroll
    for (int nt = 0; nt < 4; ++nt)
#pragma unroll
      for (int i = 0; i < 8; ++i) acc[t][nt][i] = 0.f;

  const size_t nfb = (size_t)b * NPIX * CIN;
  for (int kb = 0; kb < CIN / 32; ++kb) {
    __syncthreads();
#pragma unroll
    for (int j = 0; j < 32; ++j) {           // fill 256x32 diff chunk
      int li = j * 256 + tid;
      int er = li >> 5, kk = li & 31;
      int p0 = sP[er][0], p1 = sP[er][1];
      float a = (float)nf[nfb + (size_t)p0 * CIN + kb * 32 + kk];
      float c = (float)nf[nfb + (size_t)p1 * CIN + kb * 32 + kk];
      sD[er][kk] = (bf16)fabsf(a - c);
    }
    __syncthreads();
    // Two A frags (rows wv*32..+15 and +16..+31); lane=M, K split by lane-half
    const int r0 = wv * 32 + mrow;
    v16bf a0 = cat8(*(const v8bf*)&sD[r0][kh * 8],
                    *(const v8bf*)&sD[r0][16 + kh * 8]);
    v16bf a1 = cat8(*(const v8bf*)&sD[r0 + 16][kh * 8],
                    *(const v8bf*)&sD[r0 + 16][16 + kh * 8]);
#pragma unroll
    for (int nt = 0; nt < 4; ++nt) {
      // B frag: 32x16, lane=N, K=kh*8..+7 and 16+kh*8..+7 in sW[n][*]
      const bf16* bp = &sW[nt * 16 + mrow][kb * 32 + kh * 8];
      v16bf bfrag = cat8(*(const v8bf*)bp, *(const v8bf*)(bp + 16));
      acc[0][nt] = __builtin_amdgcn_wmma_f32_16x16x32_bf16(
          false, a0, false, bfrag, (short)0, acc[0][nt], false, false);
      acc[1][nt] = __builtin_amdgcn_wmma_f32_16x16x32_bf16(
          false, a1, false, bfrag, (short)0, acc[1][nt], false, false);
    }
  }
  const bool full = (e0 + 256) <= E;
  if (full) {
#pragma unroll
    for (int t = 0; t < 2; ++t)
#pragma unroll
      for (int nt = 0; nt < 4; ++nt) {
        int col = nt * 16 + mrow;
        float bias = b1[col];
#pragma unroll
        for (int r = 0; r < 8; ++r) {
          int e = e0 + wv * 32 + t * 16 + r + kh * 8;  // M = r + 8*(lane>=16)
          hpre[((size_t)b * E + e) * PHIH + col] = acc[t][nt][r] + bias;
        }
      }
  } else {
#pragma unroll
    for (int t = 0; t < 2; ++t)
#pragma unroll
      for (int nt = 0; nt < 4; ++nt) {
        int col = nt * 16 + mrow;
        float bias = b1[col];
#pragma unroll
        for (int r = 0; r < 8; ++r) {
          int e = e0 + wv * 32 + t * 16 + r + kh * 8;
          if (e < E) hpre[((size_t)b * E + e) * PHIH + col] = acc[t][nt][r] + bias;
        }
      }
  }
}

// ----------------------------------------------------- BN stats over edges
__global__ __launch_bounds__(256) void k_stats(const float* __restrict__ hpre,
                                               float* __restrict__ stats, int E) {
  __shared__ float ssum[256], ssq[256];
  int tid = threadIdx.x, b = blockIdx.x;
  int ch = tid & 63, g = tid >> 6;
  float s = 0.f, q = 0.f;
  for (int e = g; e < E; e += 4) {
    float x = hpre[((size_t)b * E + e) * PHIH + ch];
    s += x; q += x * x;
  }
  ssum[tid] = s; ssq[tid] = q;
  __syncthreads();
  if (tid < 64) {
    s = ssum[tid] + ssum[tid + 64] + ssum[tid + 128] + ssum[tid + 192];
    q = ssq[tid] + ssq[tid + 64] + ssq[tid + 128] + ssq[tid + 192];
    float mu = s / (float)E;
    float var = q / (float)E - mu * mu;
    stats[(b * PHIH + tid) * 2 + 0] = mu;
    stats[(b * PHIH + tid) * 2 + 1] = rsqrtf(var + BN_EPS);
  }
}

// ---------------------------------- BN + ReLU + dot(w2) + sigmoid -> edge
__global__ __launch_bounds__(256) void k_edge(const float* __restrict__ hpre,
                                              const float* __restrict__ stats,
                                              const float* __restrict__ gamma,
                                              const float* __restrict__ beta,
                                              const float* __restrict__ w2,
                                              const float* __restrict__ b2,
                                              float* __restrict__ edgev, int E) {
  __shared__ float smu[64], srs[64], sga[64], sbe[64], sw2[64];
  int tid = threadIdx.x, b = blockIdx.y;
  if (tid < 64) {
    smu[tid] = stats[(b * PHIH + tid) * 2 + 0];
    srs[tid] = stats[(b * PHIH + tid) * 2 + 1];
    sga[tid] = gamma[tid]; sbe[tid] = beta[tid]; sw2[tid] = w2[tid];
  }
  __syncthreads();
  int e = blockIdx.x * 256 + tid;
  if (e >= E) return;
  const float* hp = hpre + ((size_t)b * E + e) * PHIH;
  float acc = 0.f;
#pragma unroll
  for (int c = 0; c < PHIH; ++c) {
    float x = (hp[c] - smu[c]) * srs[c] * sga[c] + sbe[c];
    x = fmaxf(x, 0.f);
    acc += x * sw2[c];
  }
  acc += b2[0];
  edgev[(size_t)b * E + e] = 1.f / (1.f + __expf(-acc));
}

// ------------------------------------------- node GEMM: nf @ gc1 -> xw f32
// block: 256 thr; 256 rows x 128 cols; each wave owns 32 rows (two A tiles
// sharing each B fragment).
__global__ __launch_bounds__(256) void k_xw(const bf16* __restrict__ nf,
                                            const bf16* __restrict__ g1t,
                                            float* __restrict__ xw) {
  __shared__ bf16 sA[256][32];   // 16 KB
  __shared__ bf16 sB[128][32];   //  8 KB
  int tid = threadIdx.x, b = blockIdx.y;
  int row0 = blockIdx.x * 256;
  const int lane = tid & 31, wv = tid >> 5;
  const int mrow = lane & 15, kh = lane >> 4;
  v8f acc[2][8];
#pragma unroll
  for (int t = 0; t < 2; ++t)
#pragma unroll
    for (int nt = 0; nt < 8; ++nt)
#pragma unroll
      for (int i = 0; i < 8; ++i) acc[t][nt][i] = 0.f;

  const size_t nfb = (size_t)b * NPIX * CIN;
  for (int kb = 0; kb < CIN / 32; ++kb) {
    __syncthreads();
    {
      uint32_t* dA = (uint32_t*)&sA[0][0];
      uint32_t* dB = (uint32_t*)&sB[0][0];
      const uint32_t* srcn = (const uint32_t*)nf;
      const uint32_t* srcg = (const uint32_t*)g1t;
#pragma unroll
      for (int j = 0; j < 16; ++j) {         // 256 rows x 16 dwords
        int li = j * 256 + tid;
        int r = li >> 4, kp = li & 15;
        dA[li] = srcn[(nfb + (size_t)(row0 + r) * CIN + kb * 32) / 2 + kp];
      }
#pragma unroll
      for (int j = 0; j < 8; ++j) {          // 128 cols x 16 dwords
        int li = j * 256 + tid;
        int r = li >> 4, kp = li & 15;
        dB[li] = srcg[((size_t)r * CIN + kb * 32) / 2 + kp];
      }
    }
    __syncthreads();
    const int r0 = wv * 32 + mrow;
    v16bf a0 = cat8(*(const v8bf*)&sA[r0][kh * 8],
                    *(const v8bf*)&sA[r0][16 + kh * 8]);
    v16bf a1 = cat8(*(const v8bf*)&sA[r0 + 16][kh * 8],
                    *(const v8bf*)&sA[r0 + 16][16 + kh * 8]);
#pragma unroll
    for (int nt = 0; nt < 8; ++nt) {
      const bf16* bp = &sB[nt * 16 + mrow][kh * 8];
      v16bf bfrag = cat8(*(const v8bf*)bp, *(const v8bf*)(bp + 16));
      acc[0][nt] = __builtin_amdgcn_wmma_f32_16x16x32_bf16(
          false, a0, false, bfrag, (short)0, acc[0][nt], false, false);
      acc[1][nt] = __builtin_amdgcn_wmma_f32_16x16x32_bf16(
          false, a1, false, bfrag, (short)0, acc[1][nt], false, false);
    }
  }
#pragma unroll
  for (int t = 0; t < 2; ++t)
#pragma unroll
    for (int nt = 0; nt < 8; ++nt)
#pragma unroll
      for (int r = 0; r < 8; ++r) {
        int row = row0 + wv * 32 + t * 16 + r + kh * 8;
        xw[((size_t)b * NPIX + row) * GHID + nt * 16 + mrow] = acc[t][nt][r];
      }
}

// ------------------------------- sparse adj @ xw via atomics (1 wave/edge)
__global__ __launch_bounds__(256) void k_spmm1(const int* __restrict__ pairs,
                                               const float* __restrict__ edgev,
                                               const float* __restrict__ xw,
                                               float* __restrict__ h1, int E) {
  int tid = threadIdx.x, b = blockIdx.y;
  int e = blockIdx.x * 8 + (tid >> 5);
  if (e >= E) return;
  int lane = tid & 31;
  int p0 = pairs[((size_t)b * E + e) * 2 + 0];
  int p1 = pairs[((size_t)b * E + e) * 2 + 1];
  float w = edgev[(size_t)b * E + e];
  const float* src = xw + ((size_t)b * NPIX + p1) * GHID;
  float* dst = h1 + ((size_t)b * NPIX + p0) * GHID;
#pragma unroll
  for (int j = 0; j < GHID / 32; ++j)
    atomicAdd(&dst[lane + j * 32], w * src[lane + j * 32]);
}

// --------------------------------- leaky_relu(h1) . gc2_w -> hv per node
__global__ __launch_bounds__(256) void k_h1dot(const float* __restrict__ h1,
                                               const float* __restrict__ g2,
                                               float* __restrict__ hv) {
  int idx = blockIdx.x * 256 + threadIdx.x;
  if (idx >= BSZ * NPIX) return;
  const float* hp = h1 + (size_t)idx * GHID;
  float acc = 0.f;
#pragma unroll 4
  for (int d = 0; d < GHID; ++d) {
    float v = hp[d];
    v = (v > 0.f) ? v : 0.2f * v;
    acc += v * g2[d];
  }
  hv[idx] = acc;
}

// ------------------------------------------ sparse adj @ hv via atomics
__global__ __launch_bounds__(256) void k_spmm2(const int* __restrict__ pairs,
                                               const float* __restrict__ edgev,
                                               const float* __restrict__ hv,
                                               float* __restrict__ acc2, int E) {
  int b = blockIdx.y;
  int e = blockIdx.x * 256 + threadIdx.x;
  if (e >= E) return;
  int p0 = pairs[((size_t)b * E + e) * 2 + 0];
  int p1 = pairs[((size_t)b * E + e) * 2 + 1];
  atomicAdd(&acc2[b * NPIX + p0], edgev[(size_t)b * E + e] * hv[b * NPIX + p1]);
}

__global__ __launch_bounds__(256) void k_sig(const float* __restrict__ acc2,
                                             float* __restrict__ out) {
  int idx = blockIdx.x * 256 + threadIdx.x;
  if (idx >= BSZ * NPIX) return;
  out[idx] = 1.f / (1.f + __expf(-acc2[idx]));
}

// ---------------------------------------------------------------- launcher
extern "C" void kernel_launch(void* const* d_in, const int* in_sizes, int n_in,
                              void* d_out, int out_size, void* d_ws, size_t ws_size,
                              hipStream_t stream) {
  const float* sf    = (const float*)d_in[0];
  const float* xc    = (const float*)d_in[1];
  const int*   pairs = (const int*)d_in[2];
  const float* w1    = (const float*)d_in[3];
  const float* b1    = (const float*)d_in[4];
  const float* gamma = (const float*)d_in[5];
  const float* beta  = (const float*)d_in[6];
  const float* w2    = (const float*)d_in[7];
  const float* b2    = (const float*)d_in[8];
  const float* g1    = (const float*)d_in[9];
  const float* g2    = (const float*)d_in[10];
  const int E = in_sizes[2] / (2 * BSZ);
  (void)n_in; (void)out_size; (void)ws_size;

  char* ws = (char*)d_ws;
  size_t off = 0;
  auto carve = [&](size_t bytes) -> char* {
    char* p = ws + off;
    off += (bytes + 255) & ~(size_t)255;
    return p;
  };
  bf16*  nf    = (bf16*)  carve((size_t)BSZ * NPIX * CIN * sizeof(bf16));
  bf16*  w1t   = (bf16*)  carve((size_t)PHIH * CIN * sizeof(bf16));
  bf16*  g1t   = (bf16*)  carve((size_t)GHID * CIN * sizeof(bf16));
  float* hpre  = (float*) carve((size_t)BSZ * E * PHIH * sizeof(float));
  float* stats = (float*) carve((size_t)BSZ * PHIH * 2 * sizeof(float));
  float* edgev = (float*) carve((size_t)BSZ * E * sizeof(float));
  float* xw    = (float*) carve((size_t)BSZ * NPIX * GHID * sizeof(float));
  float* h1    = (float*) carve((size_t)BSZ * NPIX * GHID * sizeof(float));
  float* hv    = (float*) carve((size_t)BSZ * NPIX * sizeof(float));
  float* acc2  = (float*) carve((size_t)BSZ * NPIX * sizeof(float));

  hipMemsetAsync(h1, 0, (size_t)BSZ * NPIX * GHID * sizeof(float), stream);
  hipMemsetAsync(acc2, 0, (size_t)BSZ * NPIX * sizeof(float), stream);

  {
    int total = BSZ * NPIX * CIN;
    k_pack<<<(total + 255) / 256, 256, 0, stream>>>(sf, xc, nf);
  }
  {
    int total = PHIH * CIN + GHID * CIN;
    k_weights<<<(total + 255) / 256, 256, 0, stream>>>(w1, g1, w1t, g1t);
  }
  {
    dim3 g((E + 255) / 256, BSZ);
    k_phi<<<g, 256, 0, stream>>>(nf, w1t, pairs, b1, hpre, E);
  }
  k_stats<<<BSZ, 256, 0, stream>>>(hpre, stats, E);
  {
    dim3 g((E + 255) / 256, BSZ);
    k_edge<<<g, 256, 0, stream>>>(hpre, stats, gamma, beta, w2, b2, edgev, E);
  }
  {
    dim3 g(NPIX / 256, BSZ);
    k_xw<<<g, 256, 0, stream>>>(nf, g1t, xw);
  }
  {
    dim3 g((E + 7) / 8, BSZ);
    k_spmm1<<<g, 256, 0, stream>>>(pairs, edgev, xw, h1, E);
  }
  k_h1dot<<<(BSZ * NPIX + 255) / 256, 256, 0, stream>>>(h1, g2, hv);
  {
    dim3 g((E + 255) / 256, BSZ);
    k_spmm2<<<g, 256, 0, stream>>>(pairs, edgev, hv, acc2, E);
  }
  k_sig<<<(BSZ * NPIX + 255) / 256, 256, 0, stream>>>(acc2, (float*)d_out);
}
